// CausalSelfAttention_16827681866021
// MI455X (gfx1250) — compile-verified, matrix operand order checked by reading, not run on hardware
//
#include <hip/hip_runtime.h>

typedef __bf16 bf16;
typedef __attribute__((ext_vector_type(16))) __bf16 v16bf;
typedef __attribute__((ext_vector_type(8)))  __bf16 v8bf;
typedef __attribute__((ext_vector_type(8)))  float  v8f;
typedef __attribute__((ext_vector_type(4)))  unsigned int v4u;
typedef __attribute__((ext_vector_type(8)))  int v8i;
typedef __attribute__((ext_vector_type(4)))  int v4i;

#define SLEN   2048
#define DMODEL 1024
#define NHEADS 16
#define DHEAD  64
#define NBATCH 2

// D = A(16x32) * B(32x16) + C, bf16 inputs, f32 accumulate
static __device__ __forceinline__ v8f wmma_bf16(v16bf a, v16bf b, v8f c) {
  return __builtin_amdgcn_wmma_f32_16x16x32_bf16(false, a, false, b, (short)0, c,
                                                 false, false);
}

// Load a 16x32 bf16 fragment from row-major [16 rows, ld cols] memory.
// CDNA5 16-bit A/B layout: lane = row (mod 16); lanes 0-15 hold
// K = {0..7, 16..23}, lanes 16-31 hold K = {8..15, 24..31}.
static __device__ __forceinline__ v16bf load_frag(const bf16* __restrict__ p,
                                                  int ld, int lane) {
  const int r  = lane & 15;
  const int kb = (lane < 16) ? 0 : 8;
  const bf16* q = p + (long long)r * ld + kb;
  v8bf lo = *(const v8bf*)(q);
  v8bf hi = *(const v8bf*)(q + 16);
  return __builtin_shufflevector(lo, hi, 0, 1, 2, 3, 4, 5, 6, 7,
                                 8, 9, 10, 11, 12, 13, 14, 15);
}

// Issue a TDM 2D tile load: tile_h rows x tile_w bf16 elems, global row
// stride = stride_elems, packed contiguously into LDS at byte offset lds_off.
// D# layout per CDNA5 ISA ch.8 (group0: count/lds/global/type; group1: dims).
static __device__ __forceinline__ void
tdm_load_2d(unsigned lds_off, const bf16* g, unsigned tile_w, unsigned tile_h,
            unsigned long long stride_elems) {
  unsigned long long ga = (unsigned long long)g;
  unsigned td0 = (unsigned)stride_elems;   // tensor dim0 (row length)
  unsigned td1 = 1u << 20;                 // tensor dim1 (rows, generous)
  v4u g0;
  g0[0] = 1u;                                         // count=1, no gather
  g0[1] = lds_off;                                    // lds_addr (bytes)
  g0[2] = (unsigned)(ga & 0xffffffffu);               // global_addr[31:0]
  g0[3] = (unsigned)((ga >> 32) & 0x01ffffffu) | (2u << 30);  // [56:32]|type=2
  v8i g1;
  g1[0] = (int)(1u << 16);                            // data_size=1 (2 bytes)
  g1[1] = (int)((td0 & 0xffffu) << 16);               // abar=0 | td0[15:0]
  g1[2] = (int)(((td0 >> 16) & 0xffffu) | ((td1 & 0xffffu) << 16));
  g1[3] = (int)(((td1 >> 16) & 0xffffu) | ((tile_w & 0xffffu) << 16));
  g1[4] = (int)(tile_h & 0xffffu);                    // tile_dim1 | tile_dim2=0
  g1[5] = (int)(stride_elems & 0xffffffffu);          // dim0_stride[31:0]
  g1[6] = (int)((stride_elems >> 32) & 0xffffu);      // dim0_stride[47:32]
  g1[7] = 0;
  v4i z = {0, 0, 0, 0};
#if __clang_major__ >= 23
  v8i z8 = {0, 0, 0, 0, 0, 0, 0, 0};
  __builtin_amdgcn_tensor_load_to_lds(g0, g1, z, z, z8, 0);
#else
  __builtin_amdgcn_tensor_load_to_lds(g0, g1, z, z, 0);
#endif
}

__global__ void __launch_bounds__(256)
cvt_f32_bf16(const float* __restrict__ in, bf16* __restrict__ out, long long n) {
  long long i = (long long)blockIdx.x * blockDim.x + threadIdx.x;
  if (i < n) out[i] = (bf16)in[i];
}

// w: [K,N] f32 (row-major)  ->  wt: [N,K] bf16 (row-major) == w^T
__global__ void __launch_bounds__(256)
transpose_cvt(const float* __restrict__ w, bf16* __restrict__ wt, int K, int N) {
  long long i = (long long)blockIdx.x * blockDim.x + threadIdx.x;
  long long total = (long long)K * N;
  if (i >= total) return;
  int k = (int)(i % K);
  int n = (int)(i / K);
  wt[(long long)n * K + k] = (bf16)w[(long long)k * N + n];
}

// C[M,N] = A[M,K] * Bt[N,K]^T ; bf16 inputs, f32 accumulate.
// Block: 256 threads = 8 waves as 2(M) x 4(N); wave tile 32x64; block tile
// 64x256. A/B k-slices staged into LDS by the Tensor Data Mover, double
// buffered: wave 0 issues tensor_load_to_lds for slice t+1 while all waves
// compute slice t from LDS. Dynamic LDS only => descriptor lds_addr offsets
// start at 0.
//
// Layout (bytes): A0 [0,4K) B0 [4K,20K) A1 [20K,24K) B1 [24K,40K)
#define GA0 0u
#define GB0 4096u
#define GA1 20480u
#define GB1 24576u
#define GEMM_LDS_BYTES 40960u

template <bool F32OUT>
__global__ void __launch_bounds__(256)
gemm_bf16_wmma(const bf16* __restrict__ A, const bf16* __restrict__ Bt,
               void* __restrict__ Cout, int M, int N, int K) {
  extern __shared__ char smem[];
  const int lane = threadIdx.x & 31;
  const int wv   = threadIdx.x >> 5;
  const int wm   = wv & 1;
  const int wn   = wv >> 1;
  const int m0   = blockIdx.y * 64;
  const int n0   = blockIdx.x * 256;

  // prologue: stage k-slice 0 into buffer 0
  if (wv == 0) {
    tdm_load_2d(GA0, A  + (long long)m0 * K, 32, 64,  (unsigned long long)K);
    tdm_load_2d(GB0, Bt + (long long)n0 * K, 32, 256, (unsigned long long)K);
  }

  v8f c[2][4];
#pragma unroll
  for (int i = 0; i < 2; ++i)
#pragma unroll
    for (int t = 0; t < 4; ++t) c[i][t] = {};

  int buf = 0;
  for (int k = 0; k < K; k += 32) {
    if (wv == 0) __builtin_amdgcn_s_wait_tensorcnt(0);
    __syncthreads();
    if (wv == 0 && (k + 32) < K) {
      tdm_load_2d(buf ? GA0 : GA1, A + (long long)m0 * K + k + 32,
                  32, 64, (unsigned long long)K);
      tdm_load_2d(buf ? GB0 : GB1, Bt + (long long)n0 * K + k + 32,
                  32, 256, (unsigned long long)K);
    }
    const bf16* aT = (const bf16*)(smem + (buf ? GA1 : GA0));
    const bf16* bT = (const bf16*)(smem + (buf ? GB1 : GB0));
    v16bf a0 = load_frag(aT + (wm * 32) * 32, 32, lane);
    v16bf a1 = load_frag(aT + (wm * 32 + 16) * 32, 32, lane);
#pragma unroll
    for (int t = 0; t < 4; ++t) {
      v16bf b = load_frag(bT + (wn * 64 + 16 * t) * 32, 32, lane);
      c[0][t] = wmma_bf16(a0, b, c[0][t]);
      c[1][t] = wmma_bf16(a1, b, c[1][t]);
    }
    buf ^= 1;
  }

  // C layout: VGPR i -> row (i | i+8 per half-wave), col = lane % 16
  const int rr = (lane < 16) ? 0 : 8;
  const int cc = lane & 15;
#pragma unroll
  for (int w = 0; w < 2; ++w) {
#pragma unroll
    for (int t = 0; t < 4; ++t) {
#pragma unroll
      for (int i = 0; i < 8; ++i) {
        long long r = m0 + wm * 32 + 16 * w + rr + i;
        int col = n0 + wn * 64 + 16 * t + cc;
        if (F32OUT) ((float*)Cout)[r * N + col] = c[w][t][i];
        else        ((bf16*)Cout)[r * N + col] = (bf16)c[w][t][i];
      }
    }
  }
}

// qkv: [B*S, 3*D] bf16. Writes q,k (RoPE'd; q pre-scaled by dh^-0.5 * log2(e)
// so softmax can run in exp2 domain) as [B,H,S,dh] and v^T as [B,H,dh,S].
__global__ void __launch_bounds__(256)
rope_split(const bf16* __restrict__ qkv, bf16* __restrict__ qh,
           bf16* __restrict__ kh, bf16* __restrict__ vt) {
  int idx = blockIdx.x * blockDim.x + threadIdx.x;
  int i = idx & 31;             // rotary pair index 0..31
  int s = (idx >> 5) & (SLEN - 1);
  int h = (idx >> 16) & (NHEADS - 1);
  int b = idx >> 20;
  if (b >= NBATCH) return;

  const bf16* row = qkv + (long long)(b * SLEN + s) * (3 * DMODEL);
  float q1 = (float)row[h * DHEAD + i];
  float q2 = (float)row[h * DHEAD + 32 + i];
  float k1 = (float)row[DMODEL + h * DHEAD + i];
  float k2 = (float)row[DMODEL + h * DHEAD + 32 + i];
  float v1 = (float)row[2 * DMODEL + h * DHEAD + i];
  float v2 = (float)row[2 * DMODEL + h * DHEAD + 32 + i];

  float inv_freq = __powf(10000.0f, -(float)(2 * i) / 64.0f);
  float th = (float)s * inv_freq;
  float sn, cs;
  __sincosf(th, &sn, &cs);
  const float scale = 0.125f * 1.44269504f;  // dh^-0.5 * log2(e) folded into Q

  long long base = ((long long)(b * NHEADS + h) * SLEN + s) * DHEAD;
  qh[base + i]      = (bf16)((q1 * cs - q2 * sn) * scale);
  qh[base + 32 + i] = (bf16)((q2 * cs + q1 * sn) * scale);
  kh[base + i]      = (bf16)(k1 * cs - k2 * sn);
  kh[base + 32 + i] = (bf16)(k2 * cs + k1 * sn);

  long long vbase = (long long)(b * NHEADS + h) * DHEAD * SLEN;
  vt[vbase + (long long)i * SLEN + s]        = (bf16)v1;
  vt[vbase + (long long)(i + 32) * SLEN + s] = (bf16)v2;
}

// Process one 64-key block of flash attention for one 16-row Q tile.
// MASKED=true only for the single diagonal block.
template <bool MASKED>
static __device__ __forceinline__ void
attn_block(const bf16* __restrict__ kp, const bf16* __restrict__ vp,
           int j, int qbase, int lane, bf16 (*pw)[64],
           v16bf qf0, v16bf qf1, v16bf ones,
           v8f* o, float* mrun, float* lrun) {
  const int rr = (lane < 16) ? 0 : 8;
  const int cc = lane & 15;
  const int kb = (lane < 16) ? 0 : 8;

  // Cross-iteration prefetch of next key block (speculative, counter-free):
  // next K tile = 8KB contiguous; next V^T columns = 64 strided 128B rows.
  {
    const bf16* kn = kp + (long long)(j + 64) * DHEAD;
    __builtin_prefetch(kn + lane * 128, 0, 0);
    __builtin_prefetch(kn + lane * 128 + 64, 0, 0);
    __builtin_prefetch(vp + (long long)lane * SLEN + j + 64, 0, 0);
    __builtin_prefetch(vp + (long long)(lane + 32) * SLEN + j + 64, 0, 0);
  }

  // scores: 4 tiles of 16 keys, K-dim 64 split in two WMMAs
  v8f s[4];
#pragma unroll
  for (int t = 0; t < 4; ++t) {
    v16bf ka = load_frag(kp + (long long)(j + 16 * t) * DHEAD, DHEAD, lane);
    v16bf kbt = load_frag(kp + (long long)(j + 16 * t) * DHEAD + 32, DHEAD, lane);
    v8f acc = {};
    acc = wmma_bf16(qf0, ka, acc);
    acc = wmma_bf16(qf1, kbt, acc);
    s[t] = acc;
  }

  float alpha[8];
#pragma unroll
  for (int i = 0; i < 8; ++i) {
    int row = qbase + rr + i;
    float v0 = s[0][i], v1 = s[1][i], v2 = s[2][i], v3 = s[3][i];
    if (MASKED) {
      if (j + cc > row)      v0 = -1e30f;
      if (j + 16 + cc > row) v1 = -1e30f;
      if (j + 32 + cc > row) v2 = -1e30f;
      if (j + 48 + cc > row) v3 = -1e30f;
    }
    float rmax = fmaxf(fmaxf(v0, v1), fmaxf(v2, v3));
#pragma unroll
    for (int off = 8; off; off >>= 1)
      rmax = fmaxf(rmax, __shfl_xor(rmax, off, 32));
    float mn = fmaxf(mrun[i], rmax);
    alpha[i] = exp2f(mrun[i] - mn);
    mrun[i] = mn;
    o[0][i] *= alpha[i]; o[1][i] *= alpha[i];
    o[2][i] *= alpha[i]; o[3][i] *= alpha[i];
    pw[rr + i][cc]      = (bf16)exp2f(v0 - mn);
    pw[rr + i][16 + cc] = (bf16)exp2f(v1 - mn);
    pw[rr + i][32 + cc] = (bf16)exp2f(v2 - mn);
    pw[rr + i][48 + cc] = (bf16)exp2f(v3 - mn);
  }

  // wave-private LDS: DS ops are in-order within a wave
  asm volatile("s_wait_dscnt 0" ::: "memory");
  const int r = lane & 15;
  v8bf lo = *(const v8bf*)&pw[r][kb];
  v8bf hi = *(const v8bf*)&pw[r][kb + 16];
  v16bf pf0 = __builtin_shufflevector(lo, hi, 0, 1, 2, 3, 4, 5, 6, 7,
                                      8, 9, 10, 11, 12, 13, 14, 15);
  lo = *(const v8bf*)&pw[r][32 + kb];
  hi = *(const v8bf*)&pw[r][32 + kb + 16];
  v16bf pf1 = __builtin_shufflevector(lo, hi, 0, 1, 2, 3, 4, 5, 6, 7,
                                      8, 9, 10, 11, 12, 13, 14, 15);

  // row-sum via WMMA against all-ones B matrix (replaces shuffle reduction)
  v8f rs = {};
  rs = wmma_bf16(pf0, ones, rs);
  rs = wmma_bf16(pf1, ones, rs);
#pragma unroll
  for (int i = 0; i < 8; ++i) lrun[i] = lrun[i] * alpha[i] + rs[i];

  // O += P(16x64) @ V(64x64), V^T fragments row-major [dh, S]
#pragma unroll
  for (int t = 0; t < 4; ++t) {
    v16bf va = load_frag(vp + (long long)(16 * t) * SLEN + j, SLEN, lane);
    v16bf vb = load_frag(vp + (long long)(16 * t) * SLEN + j + 32, SLEN, lane);
    o[t] = wmma_bf16(pf0, va, o[t]);
    o[t] = wmma_bf16(pf1, vb, o[t]);
  }
}

// Flash attention. Grid: (S/64, B*H); block = 128 threads = 4 waves.
// Each wave owns one 16-row Q tile, streams keys 64 at a time.
__global__ void __launch_bounds__(128)
flash_attn(const bf16* __restrict__ Q, const bf16* __restrict__ Kk,
           const bf16* __restrict__ VT, bf16* __restrict__ O) {
  __shared__ bf16 psh[4][16][64];   // per-wave P tile staging

  const int lane  = threadIdx.x & 31;
  const int wave  = threadIdx.x >> 5;
  const int bh    = blockIdx.y;
  const int b     = bh >> 4;
  const int h     = bh & 15;
  const int qbase = (blockIdx.x * 4 + wave) * 16;

  const bf16* qp = Q  + (long long)bh * SLEN * DHEAD;
  const bf16* kp = Kk + (long long)bh * SLEN * DHEAD;
  const bf16* vp = VT + (long long)bh * DHEAD * SLEN;

  v16bf qf0 = load_frag(qp + (long long)qbase * DHEAD, DHEAD, lane);
  v16bf qf1 = load_frag(qp + (long long)qbase * DHEAD + 32, DHEAD, lane);

  v16bf ones;
#pragma unroll
  for (int i = 0; i < 16; ++i) ones[i] = (bf16)1.0f;

  v8f o[4];
#pragma unroll
  for (int t = 0; t < 4; ++t) o[t] = {};
  float mrun[8], lrun[8];
#pragma unroll
  for (int i = 0; i < 8; ++i) { mrun[i] = -1e30f; lrun[i] = 0.0f; }

  const int jtail = qbase & ~63;   // single diagonal (masked) block
  for (int j = 0; j < jtail; j += 64)
    attn_block<false>(kp, vp, j, qbase, lane, psh[wave], qf0, qf1, ones,
                      o, mrun, lrun);
  attn_block<true>(kp, vp, jtail, qbase, lane, psh[wave], qf0, qf1, ones,
                   o, mrun, lrun);

  // normalize and write [B,S,H*dh] bf16 (ready for out-projection GEMM)
  const int rr = (lane < 16) ? 0 : 8;
  const int cc = lane & 15;
  long long obase = (long long)(b * SLEN + qbase) * DMODEL + h * DHEAD;
#pragma unroll
  for (int i = 0; i < 8; ++i) {
    float inv = 1.0f / lrun[i];
    long long orow = obase + (long long)(rr + i) * DMODEL;
    O[orow + cc]      = (bf16)(o[0][i] * inv);
    O[orow + 16 + cc] = (bf16)(o[1][i] * inv);
    O[orow + 32 + cc] = (bf16)(o[2][i] * inv);
    O[orow + 48 + cc] = (bf16)(o[3][i] * inv);
  }
}

extern "C" void kernel_launch(void* const* d_in, const int* in_sizes, int n_in,
                              void* d_out, int out_size, void* d_ws, size_t ws_size,
                              hipStream_t stream) {
  const float* x     = (const float*)d_in[0];
  const float* w_qkv = (const float*)d_in[1];
  const float* w_out = (const float*)d_in[2];
  float* out = (float*)d_out;

  const long long M    = (long long)NBATCH * SLEN;            // 4096
  const long long nx   = M * DMODEL;                          // 4,194,304
  const long long nwq  = (long long)DMODEL * 3 * DMODEL;      // 3,145,728
  const long long nwo  = (long long)DMODEL * DMODEL;          // 1,048,576
  const long long nqkv = M * 3 * DMODEL;                      // 12,582,912

  bf16* xb   = (bf16*)d_ws;        // x in bf16
  bf16* wqt  = xb   + nx;          // w_qkv^T bf16 [3D, D]
  bf16* wot  = wqt  + nwq;         // w_out^T bf16 [D, D]
  bf16* qkvb = wot  + nwo;         // qkv projection bf16 [M, 3D]
  bf16* qh   = qkvb + nqkv;        // [B,H,S,dh] rope'd, scaled
  bf16* kh   = qh   + nx;          // [B,H,S,dh] rope'd
  bf16* vt   = kh   + nx;          // [B,H,dh,S]
  bf16* ob   = vt   + nx;          // attn out bf16 [M, D]

  cvt_f32_bf16<<<(int)(nx / 256), 256, 0, stream>>>(x, xb, nx);
  transpose_cvt<<<(int)(nwq / 256), 256, 0, stream>>>(w_qkv, wqt, DMODEL, 3 * DMODEL);
  transpose_cvt<<<(int)(nwo / 256), 256, 0, stream>>>(w_out, wot, DMODEL, DMODEL);

  gemm_bf16_wmma<false>
      <<<dim3(3 * DMODEL / 256, (int)(M / 64)), 256, GEMM_LDS_BYTES, stream>>>(
          xb, wqt, qkvb, (int)M, 3 * DMODEL, DMODEL);

  rope_split<<<(int)((long long)NBATCH * NHEADS * SLEN * 32 / 256), 256, 0, stream>>>(
      qkvb, qh, kh, vt);

  flash_attn<<<dim3(SLEN / 64, NBATCH * NHEADS), 128, 0, stream>>>(qh, kh, vt, ob);

  gemm_bf16_wmma<true>
      <<<dim3(DMODEL / 256, (int)(M / 64)), 256, GEMM_LDS_BYTES, stream>>>(
          ob, wot, out, (int)M, DMODEL, DMODEL);
}
